// TemporalEmbedding_15135464751440
// MI455X (gfx1250) — matrix operand order repaired and checked by reading, not compiled, not run on hardware
//
#include <hip/hip_runtime.h>

// ---------------------------------------------------------------------------
// CDNA5 / gfx1250 implementation. All heavy GEMMs go through
// v_wmma_f32_16x16x32_bf16 (bf16 operands, fp32 accumulate), with
// double-buffered GLOBAL_LOAD_ASYNC_TO_LDS_B128 staging (ASYNCcnt) so the
// DMA for tile t+1 overlaps the WMMAs on tile t.
// ---------------------------------------------------------------------------

typedef __bf16 bf16;
typedef __attribute__((ext_vector_type(16))) __bf16 v16bf;
typedef __attribute__((ext_vector_type(8)))  float  v8f;

// Problem constants (from the reference)
constexpr int       Bb   = 32;
constexpr int       NP   = 1024;
constexpr int       Cc   = 128;     // PL*CIN
constexpr int       Dm   = 512;
constexpr int       Hh   = 8;
constexpr int       DHd  = 64;
constexpr int       DFF  = 2048;
constexpr long long Tt   = (long long)Bb * NP;  // 32768 tokens
constexpr int       BHn  = Bb * Hh;             // 256 batched heads
constexpr int       Mreal = 266;                // int(64*ln(64))
constexpr int       MP    = 320;                // padded to multiple of BK=64
constexpr int       VW    = 80;                 // V width: 64 + den col + pad

// GEMM tile geometry
constexpr int GBM = 128, GBN = 64, GBK = 64, GSK = GBK + 8;

__device__ __forceinline__ void wait_asynccnt0() {
#if __has_builtin(__builtin_amdgcn_s_wait_asynccnt)
  __builtin_amdgcn_s_wait_asynccnt(0);
#else
  asm volatile("s_wait_asynccnt 0" ::: "memory");
#endif
}

// Stage one (A,B) k-tile into the given LDS buffers.
// Contiguous cases use async-to-LDS DMA (no VGPR round trip, ASYNCcnt).
template<bool TRANS_A, bool B_NK>
__device__ __forceinline__ void stage_tiles(
    bf16 (*__restrict__ sA)[GSK], bf16 (*__restrict__ sB)[GSK],
    const bf16* __restrict__ A, int lda,
    const bf16* __restrict__ B, int ldb,
    int m0, int n0, int k0, int M, int N,
    bool fullA, bool fullB, int tid)
{
  // ---- A tile (GBM x GBK) ----
  if (!TRANS_A) {
    if (fullA) {
      #pragma unroll
      for (int i = tid; i < GBM * GBK / 8; i += 256) {
        const int m = i >> 3, kk = (i & 7) * 8;
        const unsigned lp = (unsigned)(size_t)&sA[m][kk];
        const bf16* gp = A + (long long)(m0 + m) * lda + (k0 + kk);
        asm volatile("global_load_async_to_lds_b128 %0, %1, off"
                     :: "v"(lp), "v"(gp) : "memory");
      }
    } else {
      #pragma unroll
      for (int i = tid; i < GBM * GBK / 8; i += 256) {
        const int m = i >> 3, kk = (i & 7) * 8;
        uint4 val = {0u,0u,0u,0u};
        const int gr = m0 + m;
        if (gr < M) val = *(const uint4*)(A + (long long)gr * lda + (k0 + kk));
        *(uint4*)&sA[m][kk] = val;
      }
    }
  } else {
    // transpose-scatter staging (ctx GEMM): A stored (K,M)
    #pragma unroll
    for (int i = tid; i < GBM * GBK / 8; i += 256) {
      const int kk = i >> 4, m = (i & 15) * 8;   // chunk of 8 along m
      union { uint4 q; bf16 h[8]; } u; u.q = make_uint4(0u,0u,0u,0u);
      if (m0 + m < M)
        u.q = *(const uint4*)(A + (long long)(k0 + kk) * lda + (m0 + m));
      #pragma unroll
      for (int j = 0; j < 8; ++j) sA[m + j][kk] = u.h[j];
    }
  }
  // ---- B tile (GBN x GBK, stored [n][k]) ----
  if (B_NK) {
    if (fullB) {
      #pragma unroll
      for (int i = tid; i < GBN * GBK / 8; i += 256) {
        const int n = i >> 3, kk = (i & 7) * 8;
        const unsigned lp = (unsigned)(size_t)&sB[n][kk];
        const bf16* gp = B + (long long)(n0 + n) * ldb + (k0 + kk);
        asm volatile("global_load_async_to_lds_b128 %0, %1, off"
                     :: "v"(lp), "v"(gp) : "memory");
      }
    } else {
      #pragma unroll
      for (int i = tid; i < GBN * GBK / 8; i += 256) {
        const int n = i >> 3, kk = (i & 7) * 8;
        uint4 val = {0u,0u,0u,0u};
        const int gr = n0 + n;
        if (gr < N) val = *(const uint4*)(B + (long long)gr * ldb + (k0 + kk));
        *(uint4*)&sB[n][kk] = val;
      }
    }
  } else {
    #pragma unroll
    for (int i = tid; i < GBN * GBK / 8; i += 256) {
      const int kk = i >> 3, n = (i & 7) * 8;
      union { uint4 q; bf16 h[8]; } u; u.q = make_uint4(0u,0u,0u,0u);
      if (n0 + n < N)
        u.q = *(const uint4*)(B + (long long)(k0 + kk) * ldb + (n0 + n));
      #pragma unroll
      for (int j = 0; j < 8; ++j) sB[n + j][kk] = u.h[j];
    }
  }
}

// ---------------------------------------------------------------------------
// WMMA GEMM: out = A @ B' (+bias)(relu)(*rowScale)(+=)    bf16 in, f32 acc
//   A: (M,K) row-major (lda) or, TRANS_A: stored (K,M) row-major
//   B: B_NK:  W stored (N,K) row-major  -> B[k][n] = W[n*ldb+k]
//      !B_NK: W stored (K,N) row-major  -> B[k][n] = W[k*ldb+n]
// Tiles: 128x64x64, 256 threads = 8 waves, each wave 2x2 WMMA 16x16 tiles,
// 8 WMMAs per k-iteration, ping-pong LDS with async prefetch of next tile.
// ---------------------------------------------------------------------------
template<bool TRANS_A, bool B_NK, bool RELU, bool ACCUM>
__global__ __launch_bounds__(256) void gemm_bf16_wmma(
    const bf16* __restrict__ A, long long strideA, int lda,
    const bf16* __restrict__ B, long long strideB, int ldb,
    float* __restrict__ outF, bf16* __restrict__ outB, long long strideC, int ldc,
    const float* __restrict__ bias,
    const float* __restrict__ rowScale, int rsStride,
    int M, int N, int K)
{
  __shared__ __align__(16) bf16 sA[2][GBM][GSK];
  __shared__ __align__(16) bf16 sB[2][GBN][GSK];

  const int z = blockIdx.z;
  A += (long long)z * strideA;
  B += (long long)z * strideB;
  if (outF) outF += (long long)z * strideC;
  if (outB) outB += (long long)z * strideC;

  const int m0 = blockIdx.y * GBM, n0 = blockIdx.x * GBN;
  const int tid  = threadIdx.x;
  const int lane = tid & 31, wid = tid >> 5;
  const int wm = (wid >> 1) * 32, wn = (wid & 1) * 32;   // 4x2 wave grid
  const int lm = lane & 15, hl = lane >> 4;

  const bool fullA = (m0 + GBM <= M);   // uniform per block
  const bool fullB = (n0 + GBN <= N);   // uniform per block

  const v8f zf = {0.f,0.f,0.f,0.f,0.f,0.f,0.f,0.f};
  v8f acc[2][2];
  acc[0][0] = zf; acc[0][1] = zf; acc[1][0] = zf; acc[1][1] = zf;

  // prologue: stage first tile
  stage_tiles<TRANS_A, B_NK>(sA[0], sB[0], A, lda, B, ldb,
                             m0, n0, 0, M, N, fullA, fullB, tid);
  wait_asynccnt0();
  __syncthreads();

  int cur = 0;
  for (int k0 = 0; k0 < K; k0 += GBK) {
    // issue async DMA for the next tile into the back buffer
    if (k0 + GBK < K)
      stage_tiles<TRANS_A, B_NK>(sA[cur ^ 1], sB[cur ^ 1], A, lda, B, ldb,
                                 m0, n0, k0 + GBK, M, N, fullA, fullB, tid);

    // ---- compute on current buffer ----
    bf16 (*__restrict__ cA)[GSK] = sA[cur];
    bf16 (*__restrict__ cB)[GSK] = sB[cur];
    #pragma unroll
    for (int ks = 0; ks < GBK; ks += 32) {
      union { v16bf v; unsigned u[8]; } af[2], bf[2];
      #pragma unroll
      for (int t = 0; t < 2; ++t) {
        const int r = wm + t * 16 + lm;
        #pragma unroll
        for (int p = 0; p < 8; ++p) {     // A: 16-bit 16x32 K interleave
          const int kb = (p < 4 ? p * 2 : 16 + (p - 4) * 2) + hl * 8;
          af[t].u[p] = *(const unsigned*)&cA[r][ks + kb];
        }
        const int c = wn + t * 16 + lm;
        #pragma unroll
        for (int p = 0; p < 8; ++p) {     // B: 16-bit 16x32 K interleave
          const int kb = (p < 4 ? p * 2 : 16 + (p - 4) * 2) + hl * 8;
          bf[t].u[p] = *(const unsigned*)&cB[c][ks + kb];
        }
      }
      #pragma unroll
      for (int i = 0; i < 2; ++i)
        #pragma unroll
        for (int j = 0; j < 2; ++j)
          acc[i][j] = __builtin_amdgcn_wmma_f32_16x16x32_bf16(
              false, af[i].v, false, bf[j].v, (short)0, acc[i][j], false, false);
    }

    wait_asynccnt0();   // this wave's DMA into the back buffer landed
    __syncthreads();    // everyone done reading cur + back buffer visible
    cur ^= 1;
  }

  // ---- epilogue: D rows = vgpr + 8*half, cols = lane%16 ----
  #pragma unroll
  for (int i = 0; i < 2; ++i) {
    #pragma unroll
    for (int j = 0; j < 2; ++j) {
      const int c = n0 + wn + j * 16 + lm;
      if (c >= N) continue;
      const float bv = bias ? bias[c] : 0.f;
      #pragma unroll
      for (int v = 0; v < 8; ++v) {
        const int r = m0 + wm + i * 16 + hl * 8 + v;
        if (r >= M) continue;
        float val = acc[i][j][v] + bv;
        if (RELU) val = val > 0.f ? val : 0.f;
        if (rowScale) val *= rowScale[(long long)r * rsStride];
        const long long off = (long long)r * ldc + c;
        if (ACCUM)      outF[off] += val;
        else if (outF)  outF[off]  = val;
        if (outB)       outB[off]  = (bf16)val;
      }
    }
  }
}

// ---------------------------------------------------------------------------
// Elementwise / reduction kernels
// ---------------------------------------------------------------------------
__global__ void k_f32_to_bf16(const float* __restrict__ s, bf16* __restrict__ d,
                              long long n) {
  for (long long i = (long long)blockIdx.x * blockDim.x + threadIdx.x;
       i < n; i += (long long)gridDim.x * blockDim.x)
    d[i] = (bf16)s[i];
}

__global__ void k_pad_proj(const float* __restrict__ proj, bf16* __restrict__ out) {
  long long i = (long long)blockIdx.x * blockDim.x + threadIdx.x;
  if (i >= (long long)MP * DHd) return;
  const int row = (int)(i / DHd), col = (int)(i % DHd);
  out[i] = (row < Mreal) ? (bf16)proj[(long long)row * DHd + col] : (bf16)0.f;
}

// split heads: build bf16 qh,kh (scaled by dn), v_ext (ones col 64), diag terms
__global__ void k_split_heads(const float* __restrict__ q, const float* __restrict__ k,
                              const float* __restrict__ v,
                              bf16* __restrict__ qh, bf16* __restrict__ kh,
                              bf16* __restrict__ vext,
                              float* __restrict__ diagq, float* __restrict__ diagk) {
  const long long g = (long long)blockIdx.x * blockDim.x + threadIdx.x;
  if (g >= Tt * Hh) return;
  const long long t = g >> 3;
  const int h = (int)(g & 7);
  const int b = (int)(t / NP), n = (int)(t % NP);
  const long long row = ((long long)b * Hh + h) * NP + n;
  const float dn = 0.35355339059327373f;    // 64^-0.25
  float sq = 0.f, sk = 0.f;
  const long long src = t * Dm + (long long)h * DHd;
  #pragma unroll 4
  for (int d = 0; d < DHd; ++d) {
    const float qv = q[src + d], kv = k[src + d], vv = v[src + d];
    sq += qv * qv; sk += kv * kv;
    qh[row * DHd + d]  = (bf16)(qv * dn);
    kh[row * DHd + d]  = (bf16)(kv * dn);
    vext[row * VW + d] = (bf16)vv;
  }
  vext[row * VW + 64] = (bf16)1.f;
  #pragma unroll
  for (int d = 65; d < VW; ++d) vext[row * VW + d] = (bf16)0.f;
  diagq[row] = 0.0625f * sq;                // 0.5*dn*dn == 1/16 exactly
  diagk[row] = 0.0625f * sk;
}

__global__ void k_rowmax(const float* __restrict__ dash, float* __restrict__ stab,
                         long long rows) {
  const long long r = (long long)blockIdx.x * blockDim.x + threadIdx.x;
  if (r >= rows) return;
  const float* p = dash + r * MP;
  float m = p[0];
  for (int i = 1; i < Mreal; ++i) m = p[i] > m ? p[i] : m;
  stab[r] = m;
}

__device__ __forceinline__ unsigned encf(float f) {
  unsigned u = __float_as_uint(f);
  return (u & 0x80000000u) ? ~u : (u | 0x80000000u);
}
__device__ __forceinline__ float decf(unsigned u) {
  unsigned v = (u & 0x80000000u) ? (u & 0x7FFFFFFFu) : ~u;
  return __uint_as_float(v);
}

__global__ void k_gmax_init(unsigned* g) { *g = encf(-__builtin_huge_valf()); }

__global__ void k_gmax(const float* __restrict__ dash, unsigned* __restrict__ g,
                       long long rows) {
  const long long r = (long long)blockIdx.x * blockDim.x + threadIdx.x;
  if (r >= rows) return;
  const float* p = dash + r * MP;
  float m = p[0];
  for (int i = 1; i < Mreal; ++i) m = p[i] > m ? p[i] : m;
  atomicMax(g, encf(m));
}

// qp/kp = M^-0.5 * (exp(dash - diag - stab) + 1e-4), zero-padded to MP cols
__global__ void k_phi_exp(const float* __restrict__ dash, const float* __restrict__ diag,
                          const float* __restrict__ stabRow, const unsigned* __restrict__ stabG,
                          bf16* __restrict__ out, long long n) {
  const float rsM = 0.06131393394849658f;   // 266^-0.5
  for (long long i = (long long)blockIdx.x * blockDim.x + threadIdx.x;
       i < n; i += (long long)gridDim.x * blockDim.x) {
    const long long row = i / MP;
    const int m = (int)(i % MP);
    float val = 0.f;
    if (m < Mreal) {
      const float st = stabRow ? stabRow[row] : decf(*stabG);
      val = rsM * (expf(dash[i] - diag[row] - st) + 1e-4f);
    }
    out[i] = (bf16)val;
  }
}

// attn = raw[:, :64] / raw[:, 64], scattered back to (T, 512) bf16 head-concat
__global__ void k_attn_div(const float* __restrict__ raw, bf16* __restrict__ attnB,
                           long long n) {
  for (long long i = (long long)blockIdx.x * blockDim.x + threadIdx.x;
       i < n; i += (long long)gridDim.x * blockDim.x) {
    const long long bhn = i >> 6;
    const int d  = (int)(i & 63);
    const int bh = (int)(bhn / NP), nn = (int)(bhn % NP);
    const int b = bh / Hh, h = bh % Hh;
    const float den = raw[bhn * VW + 64];
    const float v   = raw[bhn * VW + d] / den;
    attnB[(((long long)b * NP + nn) * Dm) + (long long)h * DHd + d] = (bf16)v;
  }
}

// out = LN(x + y) * g + b ; optional bf16 mirror.  One wave per row (wave32).
__global__ __launch_bounds__(256) void k_add_ln(
    const float* __restrict__ x, const float* __restrict__ y,
    const float* __restrict__ g, const float* __restrict__ b,
    float* __restrict__ outF, bf16* __restrict__ outB) {
  const long long row = (long long)blockIdx.x * 8 + (threadIdx.x >> 5);
  const int lane = threadIdx.x & 31;
  const float* xr = x + row * Dm;
  const float* yr = y + row * Dm;
  float s = 0.f, ss = 0.f;
  for (int d = lane; d < Dm; d += 32) {
    const float v = xr[d] + yr[d]; s += v; ss += v * v;
  }
  #pragma unroll
  for (int o = 16; o > 0; o >>= 1) {
    s += __shfl_xor(s, o, 32); ss += __shfl_xor(ss, o, 32);
  }
  const float mu = s * (1.f / Dm);
  const float var = ss * (1.f / Dm) - mu * mu;
  const float rstd = rsqrtf(var + 1e-5f);
  for (int d = lane; d < Dm; d += 32) {
    const float v = (xr[d] + yr[d] - mu) * rstd * g[d] + b[d];
    outF[row * Dm + d] = v;
    if (outB) outB[row * Dm + d] = (bf16)v;
  }
}

// top-2-of-4 gating -> dense per-expert weights wfull (T,4)
__global__ __launch_bounds__(256) void k_gate(
    const float* __restrict__ h, const float* __restrict__ gw,
    const float* __restrict__ gb, float* __restrict__ wfull) {
  const long long t = (long long)blockIdx.x * 8 + (threadIdx.x >> 5);
  const int lane = threadIdx.x & 31;
  const float* hr = h + t * Dm;
  float a0 = 0.f, a1 = 0.f, a2 = 0.f, a3 = 0.f;
  for (int d = lane; d < Dm; d += 32) {
    const float v = hr[d];
    a0 += v * gw[0 * Dm + d]; a1 += v * gw[1 * Dm + d];
    a2 += v * gw[2 * Dm + d]; a3 += v * gw[3 * Dm + d];
  }
  #pragma unroll
  for (int o = 16; o > 0; o >>= 1) {
    a0 += __shfl_xor(a0, o, 32); a1 += __shfl_xor(a1, o, 32);
    a2 += __shfl_xor(a2, o, 32); a3 += __shfl_xor(a3, o, 32);
  }
  if (lane == 0) {
    float lg[4] = {a0 + gb[0], a1 + gb[1], a2 + gb[2], a3 + gb[3]};
    int i0 = 0;
    for (int e = 1; e < 4; ++e) if (lg[e] > lg[i0]) i0 = e;
    int i1 = -1;
    for (int e = 0; e < 4; ++e) {
      if (e == i0) continue;
      if (i1 < 0 || lg[e] > lg[i1]) i1 = e;
    }
    const float e1 = expf(lg[i1] - lg[i0]);   // stabilized softmax over top2
    const float w0 = 1.f / (1.f + e1), w1 = e1 / (1.f + e1);
    float o[4] = {0.f, 0.f, 0.f, 0.f};
    o[i0] = w0; o[i1] = w1;
    #pragma unroll
    for (int e = 0; e < 4; ++e) wfull[t * 4 + e] = o[e];
  }
}

__global__ void k_zero_f32(float* __restrict__ p, long long n) {
  for (long long i = (long long)blockIdx.x * blockDim.x + threadIdx.x;
       i < n; i += (long long)gridDim.x * blockDim.x)
    p[i] = 0.f;
}

// ---------------------------------------------------------------------------
extern "C" void kernel_launch(void* const* d_in, const int* in_sizes, int n_in,
                              void* d_out, int out_size, void* d_ws, size_t ws_size,
                              hipStream_t stream) {
  const float* x    = (const float*)d_in[0];
  const float* embw = (const float*)d_in[1];
  const float* wq   = (const float*)d_in[2];
  const float* bq   = (const float*)d_in[3];
  const float* wk   = (const float*)d_in[4];
  const float* bk   = (const float*)d_in[5];
  const float* wv   = (const float*)d_in[6];
  const float* bv   = (const float*)d_in[7];
  const float* wo   = (const float*)d_in[8];
  const float* bo   = (const float*)d_in[9];
  const float* proj = (const float*)d_in[10];
  const float* ln2g = (const float*)d_in[11];
  const float* ln2b = (const float*)d_in[12];
  const float* ln3g = (const float*)d_in[13];
  const float* ln3b = (const float*)d_in[14];
  const float* gw   = (const float*)d_in[15];
  const float* gb   = (const float*)d_in[16];
  const float* w1   = (const float*)d_in[17];
  const float* b1   = (const float*)d_in[18];
  const float* w2   = (const float*)d_in[19];
  const float* b2   = (const float*)d_in[20];

  char* wp = (char*)d_ws;
  auto alloc = [&](size_t bytes) -> void* {
    void* r = (void*)wp;
    wp += (bytes + 255) & ~(size_t)255;
    return r;
  };

  // bf16 operand buffers
  bf16* xB    = (bf16*)alloc(Tt * Cc * sizeof(bf16));
  bf16* embB  = (bf16*)alloc((size_t)Dm * Cc * sizeof(bf16));
  bf16* wqB   = (bf16*)alloc((size_t)Dm * Dm * sizeof(bf16));
  bf16* wkB   = (bf16*)alloc((size_t)Dm * Dm * sizeof(bf16));
  bf16* wvB   = (bf16*)alloc((size_t)Dm * Dm * sizeof(bf16));
  bf16* woB   = (bf16*)alloc((size_t)Dm * Dm * sizeof(bf16));
  bf16* projB = (bf16*)alloc((size_t)MP * DHd * sizeof(bf16));
  bf16* w1B   = (bf16*)alloc((size_t)4 * Dm * DFF * sizeof(bf16));
  bf16* w2B   = (bf16*)alloc((size_t)4 * DFF * Dm * sizeof(bf16));
  // activations
  float* xeF   = (float*)alloc(Tt * Dm * sizeof(float));
  bf16*  xeB   = (bf16*)alloc(Tt * Dm * sizeof(bf16));
  float* qF    = (float*)alloc(Tt * Dm * sizeof(float));
  float* kF    = (float*)alloc(Tt * Dm * sizeof(float));
  float* vF    = (float*)alloc(Tt * Dm * sizeof(float));
  bf16*  qh    = (bf16*)alloc((size_t)BHn * NP * DHd * sizeof(bf16));
  bf16*  kh    = (bf16*)alloc((size_t)BHn * NP * DHd * sizeof(bf16));
  bf16*  vext  = (bf16*)alloc((size_t)BHn * NP * VW * sizeof(bf16));
  float* diagq = (float*)alloc((size_t)BHn * NP * sizeof(float));
  float* diagk = (float*)alloc((size_t)BHn * NP * sizeof(float));
  float* stabq = (float*)alloc((size_t)BHn * NP * sizeof(float));
  unsigned* gmax = (unsigned*)alloc(256);
  float* dashQ = (float*)alloc((size_t)BHn * NP * MP * sizeof(float));
  float* dashK = (float*)alloc((size_t)BHn * NP * MP * sizeof(float));
  bf16*  qp    = (bf16*)alloc((size_t)BHn * NP * MP * sizeof(bf16));
  bf16*  kp    = (bf16*)alloc((size_t)BHn * NP * MP * sizeof(bf16));
  bf16*  ctxB  = (bf16*)alloc((size_t)BHn * MP * VW * sizeof(bf16));
  float* rawF  = (float*)alloc((size_t)BHn * NP * VW * sizeof(float));
  bf16*  attnB = (bf16*)alloc(Tt * Dm * sizeof(bf16));
  float* attoF = (float*)alloc(Tt * Dm * sizeof(float));
  float* h1F   = (float*)alloc(Tt * Dm * sizeof(float));
  bf16*  h1B   = (bf16*)alloc(Tt * Dm * sizeof(bf16));
  float* wfull = (float*)alloc(Tt * 4 * sizeof(float));
  bf16*  hmidB = (bf16*)alloc(Tt * (size_t)DFF * sizeof(bf16));
  float* moeF  = (float*)alloc(Tt * Dm * sizeof(float));

  const dim3 blk(256);
  auto cvt = [&](const float* s, bf16* d, long long n) {
    int g = (int)((n + 255) / 256); if (g > 16384) g = 16384;
    k_f32_to_bf16<<<g, blk, 0, stream>>>(s, d, n);
  };
  auto ggrid = [](int M, int N, int Z) { return dim3((N + 63) / 64, (M + 127) / 128, Z); };

  // 0) convert operands to bf16
  cvt(x, xB, Tt * Cc);
  cvt(embw, embB, (long long)Dm * Cc);
  cvt(wq, wqB, (long long)Dm * Dm);
  cvt(wk, wkB, (long long)Dm * Dm);
  cvt(wv, wvB, (long long)Dm * Dm);
  cvt(wo, woB, (long long)Dm * Dm);
  cvt(w1, w1B, (long long)4 * Dm * DFF);
  cvt(w2, w2B, (long long)4 * DFF * Dm);
  k_pad_proj<<<(MP * DHd + 255) / 256, blk, 0, stream>>>(proj, projB);

  // 1) xe = X @ emb^T                 (T,128)x(512,128)^T
  gemm_bf16_wmma<false, true, false, false><<<ggrid((int)Tt, Dm, 1), blk, 0, stream>>>(
      xB, 0, Cc, embB, 0, Cc, xeF, xeB, 0, Dm, nullptr, nullptr, 0, (int)Tt, Dm, Cc);
  // 2) q/k/v = xe @ w^T + b
  gemm_bf16_wmma<false, true, false, false><<<ggrid((int)Tt, Dm, 1), blk, 0, stream>>>(
      xeB, 0, Dm, wqB, 0, Dm, qF, nullptr, 0, Dm, bq, nullptr, 0, (int)Tt, Dm, Dm);
  gemm_bf16_wmma<false, true, false, false><<<ggrid((int)Tt, Dm, 1), blk, 0, stream>>>(
      xeB, 0, Dm, wkB, 0, Dm, kF, nullptr, 0, Dm, bk, nullptr, 0, (int)Tt, Dm, Dm);
  gemm_bf16_wmma<false, true, false, false><<<ggrid((int)Tt, Dm, 1), blk, 0, stream>>>(
      xeB, 0, Dm, wvB, 0, Dm, vF, nullptr, 0, Dm, bv, nullptr, 0, (int)Tt, Dm, Dm);
  // 3) split heads, scale by dn, build v_ext / diag
  k_split_heads<<<(int)((Tt * Hh + 255) / 256), blk, 0, stream>>>(
      qF, kF, vF, qh, kh, vext, diagq, diagk);
  // 4) dash = (q*dn) @ proj^T  (batched over 256 heads), K=64 single step
  gemm_bf16_wmma<false, true, false, false><<<ggrid(NP, MP, BHn), blk, 0, stream>>>(
      qh, (long long)NP * DHd, DHd, projB, 0, DHd, dashQ, nullptr,
      (long long)NP * MP, MP, nullptr, nullptr, 0, NP, MP, DHd);
  gemm_bf16_wmma<false, true, false, false><<<ggrid(NP, MP, BHn), blk, 0, stream>>>(
      kh, (long long)NP * DHd, DHd, projB, 0, DHd, dashK, nullptr,
      (long long)NP * MP, MP, nullptr, nullptr, 0, NP, MP, DHd);
  // 5) stabilizers: row max (q), global max (k)
  const long long rows = (long long)BHn * NP;
  k_rowmax<<<(int)((rows + 255) / 256), blk, 0, stream>>>(dashQ, stabq, rows);
  k_gmax_init<<<1, 1, 0, stream>>>(gmax);
  k_gmax<<<(int)((rows + 255) / 256), blk, 0, stream>>>(dashK, gmax, rows);
  // 6) qp/kp = phi(...)  (bf16, zero-padded to MP)
  const long long phiN = rows * MP;
  k_phi_exp<<<8192, blk, 0, stream>>>(dashQ, diagq, stabq, nullptr, qp, phiN);
  k_phi_exp<<<8192, blk, 0, stream>>>(dashK, diagk, nullptr, gmax, kp, phiN);
  // 7) ctx_ext = kp^T @ v_ext  (col 64 = ksum)   (MP,VW) per head
  gemm_bf16_wmma<true, false, false, false><<<ggrid(MP, VW, BHn), blk, 0, stream>>>(
      kp, (long long)NP * MP, MP, vext, (long long)NP * VW, VW,
      nullptr, ctxB, (long long)MP * VW, VW, nullptr, nullptr, 0, MP, VW, NP);
  // 8) raw = qp @ ctx_ext   (col 64 = den)
  gemm_bf16_wmma<false, false, false, false><<<ggrid(NP, VW, BHn), blk, 0, stream>>>(
      qp, (long long)NP * MP, MP, ctxB, (long long)MP * VW, VW,
      rawF, nullptr, (long long)NP * VW, VW, nullptr, nullptr, 0, NP, VW, MP);
  // 9) attn = raw/den, scattered to head-concat (T,512) bf16
  k_attn_div<<<8192, blk, 0, stream>>>(rawF, attnB, rows * DHd);
  // 10) attn_o = attn @ wo^T + bo
  gemm_bf16_wmma<false, true, false, false><<<ggrid((int)Tt, Dm, 1), blk, 0, stream>>>(
      attnB, 0, Dm, woB, 0, Dm, attoF, nullptr, 0, Dm, bo, nullptr, 0, (int)Tt, Dm, Dm);
  // 11) h1 = LN(xe + attn_o)
  k_add_ln<<<(int)(Tt / 8), blk, 0, stream>>>(xeF, attoF, ln2g, ln2b, h1F, h1B);
  // 12) top-2 gating
  k_gate<<<(int)(Tt / 8), blk, 0, stream>>>(h1F, gw, gb, wfull);
  // 13) MoE: dense per expert, weight folded into w2 epilogue
  k_zero_f32<<<8192, blk, 0, stream>>>(moeF, Tt * Dm);
  for (int e = 0; e < 4; ++e) {
    gemm_bf16_wmma<false, false, true, false><<<ggrid((int)Tt, DFF, 1), blk, 0, stream>>>(
        h1B, 0, Dm, w1B + (long long)e * Dm * DFF, 0, DFF,
        nullptr, hmidB, 0, DFF, b1 + (long long)e * DFF, nullptr, 0, (int)Tt, DFF, Dm);
    gemm_bf16_wmma<false, false, false, true><<<ggrid((int)Tt, Dm, 1), blk, 0, stream>>>(
        hmidB, 0, DFF, w2B + (long long)e * DFF * Dm, 0, Dm,
        moeF, nullptr, 0, Dm, b2 + (long long)e * Dm, wfull + e, 4, (int)Tt, Dm, DFF);
  }
  // 14) out = LN(h1 + moe)
  k_add_ln<<<(int)(Tt / 8), blk, 0, stream>>>(h1F, moeF, ln3g, ln3b, (float*)d_out, nullptr);
}
